// BasicBlockDy_76038101008420
// MI455X (gfx1250) — matrix operand order repaired and checked by reading, not compiled, not run on hardware
//
#include <hip/hip_runtime.h>
#include <hip/hip_bf16.h>
#include <math.h>

// ---------------------------------------------------------------------------
// Types / helpers for bf16 WMMA on gfx1250 (wave32)
// ---------------------------------------------------------------------------
typedef __attribute__((ext_vector_type(16))) __bf16 v16bf;
typedef __attribute__((ext_vector_type(8)))  __bf16 v8bf;
typedef __attribute__((ext_vector_type(8)))  float  v8f;

struct __align__(16) U4 { unsigned x, y, z, w; };
struct __align__(32) BFPair { U4 lo, hi; };

__device__ __forceinline__ v16bf frag_from(const unsigned short* p0,
                                           const unsigned short* p1) {
  BFPair t;
  t.lo = *(const U4*)p0;
  t.hi = *(const U4*)p1;
  return __builtin_bit_cast(v16bf, t);
}
// Pack 8 floats -> 8 bf16 (RNE via hardware converts) as one 16-byte chunk
__device__ __forceinline__ U4 pack8(float e0, float e1, float e2, float e3,
                                    float e4, float e5, float e6, float e7) {
  v8bf v;
  v[0] = (__bf16)e0; v[1] = (__bf16)e1; v[2] = (__bf16)e2; v[3] = (__bf16)e3;
  v[4] = (__bf16)e4; v[5] = (__bf16)e5; v[6] = (__bf16)e6; v[7] = (__bf16)e7;
  return __builtin_bit_cast(U4, v);
}
#define WMMA_BF16(A_, B_, C_) \
  __builtin_amdgcn_wmma_f32_16x16x32_bf16(false, (A_), false, (B_), (short)0, (C_), false, false)

#define BATCH 2048
#define OUT_ELEMS (2048 * 4096)       // B*C*H*W
#define ALPHA 1e-4f

// ---------------------------------------------------------------------------
// Kernel: zero BN accumulators
// ---------------------------------------------------------------------------
__global__ void k_zero(float* __restrict__ a, float* __restrict__ b) {
  int t = threadIdx.x;
  if (t < 128) { a[t] = 0.f; b[t] = 0.f; }
}

// ---------------------------------------------------------------------------
// Kernel: BN finalize (mean/var -> scale/shift)
// ---------------------------------------------------------------------------
__global__ void k_bnfin(const float* __restrict__ acc,
                        const float* __restrict__ gamma,
                        const float* __restrict__ beta,
                        float* __restrict__ ss) {
  int c = threadIdx.x;
  if (c < 64) {
    const float invn = 1.f / 131072.f;   // B*H*W
    float mean = acc[c] * invn;
    float var  = acc[64 + c] * invn - mean * mean;
    float sc   = gamma[c] * rsqrtf(var + 1e-5f);
    ss[c]      = sc;
    ss[64 + c] = beta[c] - mean * sc;
  }
}

// im2col helper: element k of padded-LDS image, for output pixel base nbase
// k = ci*9 + r ; offset within padded 10x10 image = r + 7*(r/3) = ky*10+kx
__device__ __forceinline__ float im2col_elem(const float* xpad, int ci, int r, int nbase) {
  int off = r + 7 * ((r * 11) >> 5);   // (r*11)>>5 == r/3 for r in [0,8]
  return xpad[ci * 100 + nbase + off];
}

// ---------------------------------------------------------------------------
// Kernel 1: per-sample metaconv  out1[b] = w[b](64x576) @ im2col(x[b])(576x64)
// + BN1 statistics accumulation. One block (8 waves) per sample.
// ---------------------------------------------------------------------------
__global__ __launch_bounds__(256) void k_metaconv(
    const float* __restrict__ x, const float* __restrict__ wsrc,
    float* __restrict__ out1, float* __restrict__ bnaccG) {
  __shared__ __align__(16) float          xpad[64 * 100];   // 10x10 zero-padded
  __shared__ __align__(16) unsigned short aslice[64 * 32];
  __shared__ __align__(16) unsigned short bslice[64 * 32];
  __shared__ float bnacc[128];

  const int b = blockIdx.x;
  const int t = threadIdx.x;
  const int lane = t & 31, wv = t >> 5;
  const int mtile = wv & 3, np = wv >> 2;

  const float* __restrict__ xb = x + (size_t)b * 4096;
  const float* __restrict__ wb = wsrc + (size_t)b * 36864;

  for (int i = t; i < 6400; i += 256) xpad[i] = 0.f;
  if (t < 128) bnacc[t] = 0.f;
  __syncthreads();
  for (int i = t; i < 1024; i += 256) {     // fill interior (pad offset +1,+1)
    float4 f = ((const float4*)xb)[i];
    int ci = i >> 4, p0 = (i & 15) << 2;
    int py = p0 >> 3, px = p0 & 7;
    float* d = &xpad[ci * 100 + (py + 1) * 10 + (px + 1)];
    d[0] = f.x; d[1] = f.y; d[2] = f.z; d[3] = f.w;
  }

  const int srow = t >> 2;          // 0..63 : A row / B output pixel
  const int skg  = (t & 3) << 3;    // 0,8,16,24
  const int nbase = (srow >> 3) * 10 + (srow & 7);
  const int ar  = mtile * 16 + (lane & 15);
  const int kh  = (lane >> 4) << 3;
  const int klo = (lane >> 4) << 4;
  const int n0  = (np * 2) * 16 + (lane & 15);

  v8f acc0 = {0.f,0.f,0.f,0.f,0.f,0.f,0.f,0.f};
  v8f acc1 = {0.f,0.f,0.f,0.f,0.f,0.f,0.f,0.f};

  for (int kt = 0; kt < 18; ++kt) {
    __syncthreads();
    {  // stage w slice -> bf16
      const float* s = wb + srow * 576 + kt * 32 + skg;
      float4 f0 = *(const float4*)s;
      float4 f1 = *(const float4*)(s + 4);
      *(U4*)&aslice[srow * 32 + skg] =
          pack8(f0.x, f0.y, f0.z, f0.w, f1.x, f1.y, f1.z, f1.w);
    }
    {  // stage im2col slice -> bf16
      int kk0 = kt * 32 + skg;
      int ci = (kk0 * 0x71c8) >> 18;       // kk0/9 for kk0 < 1024 (57/512 appr)
      ci = kk0 / 9;
      int r = kk0 - ci * 9;
      float e[8];
#pragma unroll
      for (int j = 0; j < 8; ++j) {
        e[j] = im2col_elem(xpad, ci, r, nbase);
        if (++r == 9) { r = 0; ++ci; }
      }
      *(U4*)&bslice[srow * 32 + skg] =
          pack8(e[0], e[1], e[2], e[3], e[4], e[5], e[6], e[7]);
    }
    __syncthreads();
    v16bf A  = frag_from(&aslice[ar * 32 + klo], &aslice[ar * 32 + klo + 8]);
    v16bf B0 = frag_from(&bslice[n0 * 32 + klo], &bslice[n0 * 32 + klo + 8]);
    v16bf B1 = frag_from(&bslice[(n0 + 16) * 32 + klo], &bslice[(n0 + 16) * 32 + klo + 8]);
    acc0 = WMMA_BF16(A, B0, acc0);
    acc1 = WMMA_BF16(A, B1, acc1);
  }

  float* __restrict__ ob = out1 + (size_t)b * 4096;
#pragma unroll
  for (int j = 0; j < 8; ++j) {
    int m = mtile * 16 + ((lane >> 4) << 3) + j;
    float v0 = acc0[j], v1 = acc1[j];
    ob[m * 64 + n0] = v0;
    ob[m * 64 + n0 + 16] = v1;
    atomicAdd(&bnacc[m], v0 + v1);
    atomicAdd(&bnacc[64 + m], v0 * v0 + v1 * v1);
  }
  __syncthreads();
  if (t < 128) atomicAdd(&bnaccG[t], bnacc[t]);
}

// ---------------------------------------------------------------------------
// Kernel 2: out2[b] = conv2_w(64x576) @ im2col(relu(bn1(out1[b])))(576x64)
// conv2_w resident in LDS as bf16; BN2 statistics accumulation.
// ---------------------------------------------------------------------------
__global__ __launch_bounds__(256) void k_conv2(
    const float* __restrict__ out1, const float* __restrict__ c2w,
    const float* __restrict__ bn1ss, float* __restrict__ out2,
    float* __restrict__ bnaccG) {
  __shared__ __align__(16) unsigned short alds[64 * 576];
  __shared__ __align__(16) float          xpad[64 * 100];
  __shared__ __align__(16) unsigned short bslice[64 * 32];
  __shared__ float bnacc[128];

  const int b = blockIdx.x;
  const int t = threadIdx.x;
  const int lane = t & 31, wv = t >> 5;
  const int mtile = wv & 3, np = wv >> 2;

  // conv2_w -> bf16 LDS (full 64x576)
  for (int i = t; i < 4608; i += 256) {
    const float* s = c2w + i * 8;
    float4 f0 = *(const float4*)s;
    float4 f1 = *(const float4*)(s + 4);
    *(U4*)&alds[i * 8] = pack8(f0.x, f0.y, f0.z, f0.w, f1.x, f1.y, f1.z, f1.w);
  }
  for (int i = t; i < 6400; i += 256) xpad[i] = 0.f;
  if (t < 128) bnacc[t] = 0.f;
  __syncthreads();
  // relu(bn1(out1)) -> padded LDS
  const float* __restrict__ o1b = out1 + (size_t)b * 4096;
  for (int i = t; i < 1024; i += 256) {
    int ci = i >> 4;
    float sc = bn1ss[ci], sh = bn1ss[64 + ci];
    float4 f = ((const float4*)o1b)[i];
    int p0 = (i & 15) << 2;
    int py = p0 >> 3, px = p0 & 7;
    float* d = &xpad[ci * 100 + (py + 1) * 10 + (px + 1)];
    d[0] = fmaxf(fmaf(f.x, sc, sh), 0.f);
    d[1] = fmaxf(fmaf(f.y, sc, sh), 0.f);
    d[2] = fmaxf(fmaf(f.z, sc, sh), 0.f);
    d[3] = fmaxf(fmaf(f.w, sc, sh), 0.f);
  }

  const int srow = t >> 2;
  const int skg  = (t & 3) << 3;
  const int nbase = (srow >> 3) * 10 + (srow & 7);
  const int ar  = mtile * 16 + (lane & 15);
  const int kh  = (lane >> 4) << 3;
  const int klo = (lane >> 4) << 4;
  const int n0  = (np * 2) * 16 + (lane & 15);

  v8f acc0 = {0.f,0.f,0.f,0.f,0.f,0.f,0.f,0.f};
  v8f acc1 = {0.f,0.f,0.f,0.f,0.f,0.f,0.f,0.f};

  for (int kt = 0; kt < 18; ++kt) {
    __syncthreads();
    {
      int kk0 = kt * 32 + skg;
      int ci = kk0 / 9;
      int r = kk0 - ci * 9;
      float e[8];
#pragma unroll
      for (int j = 0; j < 8; ++j) {
        e[j] = im2col_elem(xpad, ci, r, nbase);
        if (++r == 9) { r = 0; ++ci; }
      }
      *(U4*)&bslice[srow * 32 + skg] =
          pack8(e[0], e[1], e[2], e[3], e[4], e[5], e[6], e[7]);
    }
    __syncthreads();
    v16bf A  = frag_from(&alds[ar * 576 + kt * 32 + klo],
                         &alds[ar * 576 + kt * 32 + klo + 8]);
    v16bf B0 = frag_from(&bslice[n0 * 32 + klo], &bslice[n0 * 32 + klo + 8]);
    v16bf B1 = frag_from(&bslice[(n0 + 16) * 32 + klo], &bslice[(n0 + 16) * 32 + klo + 8]);
    acc0 = WMMA_BF16(A, B0, acc0);
    acc1 = WMMA_BF16(A, B1, acc1);
  }

  float* __restrict__ ob = out2 + (size_t)b * 4096;
#pragma unroll
  for (int j = 0; j < 8; ++j) {
    int m = mtile * 16 + ((lane >> 4) << 3) + j;
    float v0 = acc0[j], v1 = acc1[j];
    ob[m * 64 + n0] = v0;
    ob[m * 64 + n0 + 16] = v1;
    atomicAdd(&bnacc[m], v0 + v1);
    atomicAdd(&bnacc[64 + m], v0 * v0 + v1 * v1);
  }
  __syncthreads();
  if (t < 128) atomicAdd(&bnaccG[t], bnacc[t]);
}

// ---------------------------------------------------------------------------
// Kernel 3: tail — bn2 apply, (last loop: residual+relu), pooling,
// dw = h_bar(64x32) @ lin_w^T(32x576), total_w += alpha*tanh(dw + lin_b)
// ---------------------------------------------------------------------------
__global__ __launch_bounds__(256) void k_tail(
    const float* __restrict__ x, const float* __restrict__ out2,
    const float* __restrict__ bn2ss, const float* __restrict__ lin_w,
    const float* __restrict__ lin_b, const float* __restrict__ twin,
    float* __restrict__ twout, float* __restrict__ outF, int write_final) {
  __shared__ __align__(16) float          ylds[4096];
  __shared__ __align__(16) unsigned short hbar[64 * 32];
  __shared__ __align__(16) unsigned short blds[576 * 32];

  const int b = blockIdx.x;
  const int t = threadIdx.x;
  const int lane = t & 31, wv = t >> 5;
  const int mtile = wv & 3, np = wv >> 2;

  // bn2 apply (+ fused final residual relu on last loop)
  const float* __restrict__ o2b = out2 + (size_t)b * 4096;
  for (int i = t; i < 1024; i += 256) {
    int ci = i >> 4;
    float sc = bn2ss[ci], sh = bn2ss[64 + ci];
    float4 f = ((const float4*)o2b)[i];
    f.x = fmaf(f.x, sc, sh); f.y = fmaf(f.y, sc, sh);
    f.z = fmaf(f.z, sc, sh); f.w = fmaf(f.w, sc, sh);
    ((float4*)ylds)[i] = f;
    if (write_final) {
      float4 xi = ((const float4*)(x + (size_t)b * 4096))[i];
      float4 o;
      o.x = fmaxf(f.x + xi.x, 0.f); o.y = fmaxf(f.y + xi.y, 0.f);
      o.z = fmaxf(f.z + xi.z, 0.f); o.w = fmaxf(f.w + xi.w, 0.f);
      ((float4*)(outF + (size_t)b * 4096))[i] = o;
    }
  }
  // lin_w [576,32] -> bf16 LDS, row-major (exactly the B layout we need)
  for (int i = t; i < 2304; i += 256) {
    const float* s = lin_w + i * 8;
    float4 f0 = *(const float4*)s;
    float4 f1 = *(const float4*)(s + 4);
    *(U4*)&blds[i * 8] = pack8(f0.x, f0.y, f0.z, f0.w, f1.x, f1.y, f1.z, f1.w);
  }
  __syncthreads();

  // pooling: h_bar[k][c], k=0..63, c=0..31; ch2 = 2k + c/16 (avg<64, max>=64)
  {
    int k  = t >> 2;
    int c0 = (t & 3) << 3;
    float e[8];
#pragma unroll
    for (int j = 0; j < 8; ++j) {
      int c = c0 + j;
      int ch2 = 2 * k + (c >> 4);
      int s = c & 15;
      int oy = s >> 2, ox = s & 3;
      float val;
      if (ch2 < 64) {
        float sum = 0.f;
        for (int dy = -2; dy <= 2; ++dy)
          for (int dx = -2; dx <= 2; ++dx) {
            int iy = oy * 2 + dy, ix = ox * 2 + dx;
            if (iy >= 0 && iy < 8 && ix >= 0 && ix < 8)
              sum += ylds[ch2 * 64 + iy * 8 + ix];
          }
        val = sum * 0.04f;  // count_include_pad: /25
      } else {
        int ch = ch2 - 64;
        float mx = -3.4e38f;
        for (int dy = -2; dy <= 2; ++dy)
          for (int dx = -2; dx <= 2; ++dx) {
            int iy = oy * 2 + dy, ix = ox * 2 + dx;
            if (iy >= 0 && iy < 8 && ix >= 0 && ix < 8)
              mx = fmaxf(mx, ylds[ch * 64 + iy * 8 + ix]);
          }
        val = mx;
      }
      e[j] = val;
    }
    *(U4*)&hbar[t * 8] = pack8(e[0], e[1], e[2], e[3], e[4], e[5], e[6], e[7]);
  }
  __syncthreads();

  // dw GEMM: M=64 (4 tiles), N=576 (36 tiles), K=32 (one WMMA each)
  const int ar  = mtile * 16 + (lane & 15);
  const int kh  = (lane >> 4) << 3;
  const int klo = (lane >> 4) << 4;
  v16bf A = frag_from(&hbar[ar * 32 + klo], &hbar[ar * 32 + klo + 8]);
  const size_t base = (size_t)b * 36864;

  for (int it = 0; it < 18; ++it) {
    int nt = np + 2 * it;               // covers 0..35 across the two wave groups
    int n = nt * 16 + (lane & 15);
    v16bf Bf = frag_from(&blds[n * 32 + klo], &blds[n * 32 + klo + 8]);
    v8f acc = {0.f,0.f,0.f,0.f,0.f,0.f,0.f,0.f};
    acc = WMMA_BF16(A, Bf, acc);
    float lb = lin_b[n];
#pragma unroll
    for (int j = 0; j < 8; ++j) {
      int m = mtile * 16 + ((lane >> 4) << 3) + j;
      size_t idx = base + (size_t)m * 576 + (size_t)n;
      twout[idx] = twin[idx] + ALPHA * tanhf(acc[j] + lb);
    }
  }
}

// ---------------------------------------------------------------------------
// Host launcher
// ---------------------------------------------------------------------------
extern "C" void kernel_launch(void* const* d_in, const int* in_sizes, int n_in,
                              void* d_out, int out_size, void* d_ws, size_t ws_size,
                              hipStream_t stream) {
  const float* x       = (const float*)d_in[0];
  const float* w       = (const float*)d_in[1];
  const float* conv2_w = (const float*)d_in[2];
  const float* g1      = (const float*)d_in[3];
  const float* be1     = (const float*)d_in[4];
  const float* g2      = (const float*)d_in[5];
  const float* be2     = (const float*)d_in[6];
  const float* lin_w   = (const float*)d_in[7];
  const float* lin_b   = (const float*)d_in[8];

  float* outF = (float*)d_out;            // [B,64,8,8]
  float* totw = outF + (size_t)OUT_ELEMS; // [B,64,64,3,3] lives directly in d_out

  float* ws     = (float*)d_ws;
  float* out1   = ws;
  float* out2   = ws + (size_t)OUT_ELEMS;
  float* bn1acc = ws + (size_t)2 * OUT_ELEMS;
  float* bn1ss  = bn1acc + 128;
  float* bn2acc = bn1ss + 128;
  float* bn2ss  = bn2acc + 128;

  for (int loop = 0; loop < 3; ++loop) {
    k_zero<<<1, 128, 0, stream>>>(bn1acc, bn2acc);
    const float* wsrc = (loop == 0) ? w : (const float*)totw;
    k_metaconv<<<BATCH, 256, 0, stream>>>(x, wsrc, out1, bn1acc);
    k_bnfin<<<1, 64, 0, stream>>>(bn1acc, g1, be1, bn1ss);
    k_conv2<<<BATCH, 256, 0, stream>>>(out1, conv2_w, bn1ss, out2, bn2acc);
    k_bnfin<<<1, 64, 0, stream>>>(bn2acc, g2, be2, bn2ss);
    k_tail<<<BATCH, 256, 0, stream>>>(x, out2, bn2ss, lin_w, lin_b,
                                      (loop == 0) ? w : (const float*)totw,
                                      totw, outF, (loop == 2) ? 1 : 0);
  }
}
